// VoxelContrastiveLoss_4879082848836
// MI455X (gfx1250) — compile-verified, matrix operand order checked by reading, not run on hardware
//
#include <hip/hip_runtime.h>

// ---------------- types ----------------
typedef __bf16 bf16_t;
typedef __attribute__((ext_vector_type(16))) __bf16 v16bf;
typedef __attribute__((ext_vector_type(8)))  __bf16 v8bf;
typedef __attribute__((ext_vector_type(8)))  float  v8f;
typedef __attribute__((ext_vector_type(4))) unsigned int u32x4;
typedef __attribute__((ext_vector_type(8))) int i32x8;
typedef __attribute__((ext_vector_type(4))) int i32x4;

#define NTEACH 8192
#define NSTUD  8192
#define FD     128
#define TAUF   0.07f
#define KTOP   2457          // max(1, int(8192*0.3))
#define NBINS  256           // histogram bins over [-1,1]

// WMMA 16x16x32 bf16 fragment (ISA 7.12.2): lane L, m=L%16, g=L/16.
// Element e: K = k0 + 8g + e (e<8), K = k0 + 16 + 8g + (e-8) (e>=8)
// => per lane the fragment is two contiguous 16B chunks of one row.
__device__ __forceinline__ v16bf load_frag2(const bf16_t* p /* row + k0 + 8g */) {
    v8bf lo = *(const v8bf*)(p);
    v8bf hi = *(const v8bf*)(p + 16);
    return __builtin_shufflevector(lo, hi, 0,1,2,3,4,5,6,7,8,9,10,11,12,13,14,15);
}

__device__ __forceinline__ v16bf load_frag_rm(const bf16_t* P, int ld, int r0, int k0) {
    int lane = threadIdx.x & 31;
    int m = lane & 15, g = lane >> 4;
    return load_frag2(P + (size_t)(r0 + m) * ld + k0 + 8 * g);
}

// One 16x16 sim tile vs teacher rows [ctl*16, ctl*16+16) of the staged chunk.
// All 8 ds_load_b128 issued before the WMMA chain (single dscnt wait).
__device__ __forceinline__ v8f sim_tile(const bf16_t* cur, int ctl, int m, int g,
                                        const v16bf* afr) {
    const bf16_t* base = &cur[(ctl * 16 + m) * FD + 8 * g];
    v16bf b0 = load_frag2(base);
    v16bf b1 = load_frag2(base + 32);
    v16bf b2 = load_frag2(base + 64);
    v16bf b3 = load_frag2(base + 96);
    v8f c = {};
    c = __builtin_amdgcn_wmma_f32_16x16x32_bf16(false, afr[0], false, b0, (short)0, c, false, false);
    c = __builtin_amdgcn_wmma_f32_16x16x32_bf16(false, afr[1], false, b1, (short)0, c, false, false);
    c = __builtin_amdgcn_wmma_f32_16x16x32_bf16(false, afr[2], false, b2, (short)0, c, false, false);
    c = __builtin_amdgcn_wmma_f32_16x16x32_bf16(false, afr[3], false, b3, (short)0, c, false, false);
    return c;
}

// ---------------- CDNA5 async global->LDS (per-lane, ASYNCcnt) ----------------
__device__ __forceinline__ void stage_tile_async(const bf16_t* gsrc, bf16_t* dst_lds) {
#pragma unroll
    for (int i = 0; i < 4; ++i) {
        int piece = threadIdx.x + i * 256;                           // 1024 x 16B
        unsigned loff = (unsigned)(uintptr_t)(dst_lds + piece * 8);  // low 32b = LDS offset
        const bf16_t* g = gsrc + piece * 8;
        asm volatile("global_load_async_to_lds_b128 %0, %1, off"
                     :: "v"(loff), "v"(g) : "memory");
    }
}

// ---------------- CDNA5 Tensor Data Mover (TENSORcnt) ----------------
// D# per ISA 8.3-8.5: 2D tensor (128 x 8192 bf16), tile 128 x 64 -> 16KB to LDS.
__device__ __forceinline__ void tdm_load_chunk(const bf16_t* gsrc, bf16_t* dst_lds) {
    unsigned lds = (unsigned)(uintptr_t)dst_lds;
    unsigned long long ga = (unsigned long long)(uintptr_t)gsrc;
    u32x4 g0;
    g0[0] = 1u;                                               // count=1, user descriptor
    g0[1] = lds;                                              // lds_addr
    g0[2] = (unsigned)(ga & 0xffffffffu);                     // global_addr[31:0]
    g0[3] = (unsigned)((ga >> 32) & 0x1ffffffu) | (2u << 30); // addr[56:32] | type=2
    i32x8 g1;
    g1[0] = (int)(1u << 16);       // workgroup_mask=0 | data_size=1 (2 bytes)
    g1[1] = (int)(128u << 16);     // tensor_dim0[15:0]=128 (elements per row)
    g1[2] = (int)(8192u << 16);    // tensor_dim0[31:16]=0 | tensor_dim1[15:0]=8192
    g1[3] = (int)(128u << 16);     // tensor_dim1[31:16]=0 | tile_dim0=128
    g1[4] = (int)64;               // tile_dim1=64 | tile_dim2=0
    g1[5] = (int)128;              // tensor_dim0_stride[31:0]=128
    g1[6] = 0;
    g1[7] = 0;
    i32x4 gz4 = {0, 0, 0, 0};      // groups 2/3 unused (2D tensor)
    i32x8 gz8 = {0, 0, 0, 0, 0, 0, 0, 0};
    __builtin_amdgcn_tensor_load_to_lds(g0, g1, gz4, gz4, gz8, 0);
}

// ---------------- small utility kernels ----------------
__global__ void k_init_accum(float* accum) {
    if (threadIdx.x < 2) accum[threadIdx.x] = 0.0f;
}

__global__ void k_cvt_bf16(const float* __restrict__ in, bf16_t* __restrict__ out, int n) {
    int i = blockIdx.x * blockDim.x + threadIdx.x;
    if (i < n) out[i] = (bf16_t)in[i];
}

__global__ void k_cvt_bf16_t(const float* __restrict__ in, bf16_t* __restrict__ out,
                             int R, int C) {
    int i = blockIdx.x * blockDim.x + threadIdx.x;
    if (i < R * C) {
        int r = i / C, c = i % C;
        out[(size_t)c * R + r] = (bf16_t)in[i];
    }
}

// ---------------- MLP GEMM: out = act(A[Mx128] @ W[128x128] + bias) ----------------
template <bool RELU, bool F32OUT>
__global__ __launch_bounds__(256) void k_gemm_mlp(const bf16_t* __restrict__ A,
                                                  const bf16_t* __restrict__ Wt,
                                                  const float* __restrict__ bias,
                                                  float* __restrict__ outF,
                                                  bf16_t* __restrict__ outB) {
    int wave = threadIdx.x >> 5;
    int lane = threadIdx.x & 31;
    int tile = blockIdx.x * 8 + wave;      // (M/16)*8 tiles
    int tM = tile >> 3, tN = tile & 7;     // N/16 == 8
    v8f c = {};
#pragma unroll
    for (int k0 = 0; k0 < FD; k0 += 32) {
        v16bf a = load_frag_rm(A,  FD, tM * 16, k0);
        v16bf b = load_frag_rm(Wt, FD, tN * 16, k0);
        c = __builtin_amdgcn_wmma_f32_16x16x32_bf16(false, a, false, b, (short)0, c, false, false);
    }
    int n = tN * 16 + (lane & 15);
    float bv = bias[n];
    int m0 = tM * 16 + 8 * (lane >> 4);    // D layout: VGPR r -> M = r + 8*(lane/16)
#pragma unroll
    for (int r = 0; r < 8; ++r) {
        float v = c[r] + bv;
        if (RELU) v = fmaxf(v, 0.0f);
        size_t idx = (size_t)(m0 + r) * FD + n;
        if (F32OUT) outF[idx] = v;
        else        outB[idx] = (bf16_t)v;
    }
}

// ---------------- BatchNorm stats ----------------
__global__ __launch_bounds__(256) void k_bn_stats(const float* __restrict__ H,
                                                  float* __restrict__ mu,
                                                  float* __restrict__ var, int M) {
    int c = blockIdx.x;
    float s = 0.f, s2 = 0.f;
    for (int r = threadIdx.x; r < M; r += 256) {
        float v = H[(size_t)r * FD + c];
        s += v; s2 += v * v;
    }
    __shared__ float bs[256], bs2[256];
    bs[threadIdx.x] = s; bs2[threadIdx.x] = s2; __syncthreads();
    for (int off = 128; off > 0; off >>= 1) {
        if (threadIdx.x < off) { bs[threadIdx.x] += bs[threadIdx.x + off]; bs2[threadIdx.x] += bs2[threadIdx.x + off]; }
        __syncthreads();
    }
    if (threadIdx.x == 0) {
        float m = bs[0] / (float)M;
        mu[c] = m;
        var[c] = bs2[0] / (float)M - m * m;   // ddof=0
    }
}

// ---------------- BN apply + row L2 normalize ----------------
__global__ __launch_bounds__(128) void k_bn_norm(const float* __restrict__ H,
                                                 const float* __restrict__ mu,
                                                 const float* __restrict__ var,
                                                 const float* __restrict__ gamma,
                                                 const float* __restrict__ beta,
                                                 float* __restrict__ projF,
                                                 bf16_t* __restrict__ projB) {
    int row = blockIdx.x, c = threadIdx.x;
    float v = (H[(size_t)row * FD + c] - mu[c]) * rsqrtf(var[c] + 1e-5f) * gamma[c] + beta[c];
    __shared__ float sh[128];
    sh[c] = v * v; __syncthreads();
    for (int off = 64; off > 0; off >>= 1) {
        if (c < off) sh[c] += sh[c + off];
        __syncthreads();
    }
    float o = v * rsqrtf(sh[0]);
    projF[(size_t)row * FD + c] = o;
    projB[(size_t)row * FD + c] = (bf16_t)o;
}

// ---------------- coord keys ----------------
__global__ void k_keys(const int* __restrict__ coords, int* __restrict__ keys, int n) {
    int i = blockIdx.x * blockDim.x + threadIdx.x;
    if (i < n) {
        const int* c = coords + (size_t)i * 4;
        keys[i] = ((c[0] * 64 + c[1]) * 128 + c[2]) * 128 + c[3];  // GRID_Z=64,Y=128,X=128
    }
}

// ---------------- match + positive similarity (raw dot) ----------------
__global__ __launch_bounds__(256) void k_match(const int* __restrict__ scoords,
                                               const int* __restrict__ tkeys,
                                               const float* __restrict__ sproj,
                                               const float* __restrict__ tproj,
                                               float* __restrict__ pos_raw,
                                               float* __restrict__ mflag,
                                               float* __restrict__ accum) {
    int i = blockIdx.x * blockDim.x + threadIdx.x;
    if (i >= NSTUD) return;
    const int* c = scoords + (size_t)i * 4;
    int key = ((c[0] * 64 + c[1]) * 128 + c[2]) * 128 + c[3];
    int idx = 0, found = 0;
    for (int j = 0; j < NTEACH; ++j) {       // first match == argmax of bool row
        if (tkeys[j] == key) { idx = j; found = 1; break; }
    }
    float d = 0.f;
    const float* sp = sproj + (size_t)i * FD;
    const float* tp = tproj + (size_t)idx * FD;
    for (int k = 0; k < FD; ++k) d += sp[k] * tp[k];
    pos_raw[i] = d;
    mflag[i] = (float)found;
    if (found) atomicAdd(&accum[1], 1.0f);
}

// ---------------- fused GEMM + top-k logsumexp ----------------
__global__ __launch_bounds__(256) void k_topk_lse(const bf16_t* __restrict__ sB,
                                                  const bf16_t* __restrict__ tB,
                                                  const float* __restrict__ pos_raw,
                                                  const float* __restrict__ mflag,
                                                  float* __restrict__ accum) {
    __shared__ __align__(16) bf16_t sA[64 * FD];        // 16 KB student tile
    __shared__ __align__(16) bf16_t sT0[64 * FD];       // 16 KB teacher chunk buf 0
    __shared__ __align__(16) bf16_t sT1[64 * FD];       // 16 KB teacher chunk buf 1
    __shared__ unsigned hist[64 * NBINS];               // 64 KB
    __shared__ float sumAbove[64], sumIn[64], scaleF[64];
    __shared__ int   thrBin[64];

    int wg0 = blockIdx.x * 64;
    stage_tile_async(sB + (size_t)wg0 * FD, sA);        // ASYNCcnt path
    for (int t = threadIdx.x; t < 64 * NBINS; t += 256) hist[t] = 0u;
    if (threadIdx.x < 64) { sumAbove[threadIdx.x] = 0.f; sumIn[threadIdx.x] = 0.f; }
    asm volatile("s_wait_asynccnt 0x0" ::: "memory");
    __syncthreads();

    int wave = threadIdx.x >> 5, lane = threadIdx.x & 31;
    int rt = wave >> 1;            // row tile 0..3 (16 rows each)
    int phase = wave & 1;          // column-tile interleave within a chunk
    int m = lane & 15, g = lane >> 4;
    int lr0 = rt * 16 + 8 * g;     // first of this lane's 8 result rows
    bool wave0 = (__builtin_amdgcn_readfirstlane(threadIdx.x) < 32); // scalar branch

    // Preload this wave's 4 A K-fragments from LDS (ds_load_b128 pairs).
    v16bf afr[4];
#pragma unroll
    for (int kk = 0; kk < 4; ++kk)
        afr[kk] = load_frag2(&sA[(rt * 16 + m) * FD + kk * 32 + 8 * g]);

    const int CHUNKS = NTEACH / 64;

    // ---- Pass 0: histogram of sims over [-1,1] ----
    if (wave0) tdm_load_chunk(tB, sT0);
    for (int ch = 0; ch < CHUNKS; ++ch) {
        bf16_t* cur = (ch & 1) ? sT1 : sT0;
        bf16_t* nxt = (ch & 1) ? sT0 : sT1;
        if (wave0) {
            if (ch + 1 < CHUNKS) {
                tdm_load_chunk(tB + (size_t)(ch + 1) * 64 * FD, nxt);
                __builtin_amdgcn_s_wait_tensorcnt(1);
            } else {
                __builtin_amdgcn_s_wait_tensorcnt(0);
            }
        }
        __syncthreads();
#pragma unroll
        for (int i = 0; i < 2; ++i) {                   // uniform trip count
            int ctl = phase + 2 * i;
            v8f c = sim_tile(cur, ctl, m, g, afr);
#pragma unroll
            for (int r = 0; r < 8; ++r) {
                int bin = (int)((c[r] + 1.0f) * (NBINS * 0.5f));
                bin = bin < 0 ? 0 : (bin > NBINS - 1 ? NBINS - 1 : bin);
                atomicAdd(&hist[(lr0 + r) * NBINS + bin], 1u);      // ds_add
            }
        }
        __syncthreads();
    }

    // ---- scan: k-th-value threshold bin per row ----
    if (threadIdx.x < 64) {
        int row = threadIdx.x;
        unsigned cum = 0; int t = 0; float sc = 0.f;
        for (int b = NBINS - 1; b >= 0; --b) {
            unsigned h = hist[row * NBINS + b];
            if (cum + h >= (unsigned)KTOP) {
                t = b;
                sc = h ? (float)(KTOP - cum) / (float)h : 0.f;  // pro-rate threshold bin
                break;
            }
            cum += h;
        }
        thrBin[row] = t; scaleF[row] = sc;
    }
    __syncthreads();

    // ---- Pass 1: register-accumulated exp sums, branchless select ----
    int t8[8];
#pragma unroll
    for (int r = 0; r < 8; ++r) t8[r] = thrBin[lr0 + r];
    float accA[8], accI[8];
#pragma unroll
    for (int r = 0; r < 8; ++r) { accA[r] = 0.f; accI[r] = 0.f; }

    if (wave0) tdm_load_chunk(tB, sT0);
    for (int ch = 0; ch < CHUNKS; ++ch) {
        bf16_t* cur = (ch & 1) ? sT1 : sT0;
        bf16_t* nxt = (ch & 1) ? sT0 : sT1;
        if (wave0) {
            if (ch + 1 < CHUNKS) {
                tdm_load_chunk(tB + (size_t)(ch + 1) * 64 * FD, nxt);
                __builtin_amdgcn_s_wait_tensorcnt(1);
            } else {
                __builtin_amdgcn_s_wait_tensorcnt(0);
            }
        }
        __syncthreads();
#pragma unroll
        for (int i = 0; i < 2; ++i) {
            int ctl = phase + 2 * i;
            v8f c = sim_tile(cur, ctl, m, g, afr);
#pragma unroll
            for (int r = 0; r < 8; ++r) {
                float v = c[r];
                int bin = (int)((v + 1.0f) * (NBINS * 0.5f));
                bin = bin < 0 ? 0 : (bin > NBINS - 1 ? NBINS - 1 : bin);
                float e = __expf(v * (1.0f / TAUF));    // exp(1/tau) ~ 1.6e6, no overflow
                accA[r] += (bin >  t8[r]) ? e : 0.f;    // cndmask, no branches
                accI[r] += (bin == t8[r]) ? e : 0.f;
            }
        }
        __syncthreads();
    }
#pragma unroll
    for (int r = 0; r < 8; ++r) {
        atomicAdd(&sumAbove[lr0 + r], accA[r]);         // 16 ds_add_f32 per lane total
        atomicAdd(&sumIn[lr0 + r],    accI[r]);
    }
    __syncthreads();

    if (threadIdx.x < 64) {
        int row = threadIdx.x, gr = wg0 + row;
        float sumTop = sumAbove[row] + sumIn[row] * scaleF[row];
        float p = pos_raw[gr] * (1.0f / TAUF);
        float pp = __logf(__expf(p) + sumTop) - p;   // -log_softmax[:,0]
        if (mflag[gr] > 0.0f) atomicAdd(&accum[0], pp);
    }
}

// ---------------- finalize ----------------
__global__ void k_finalize(const float* __restrict__ accum, float* __restrict__ out) {
    if (threadIdx.x == 0) out[0] = accum[0] / fmaxf(accum[1], 1.0f);
}

// ---------------- host launch ----------------
extern "C" void kernel_launch(void* const* d_in, const int* in_sizes, int n_in,
                              void* d_out, int out_size, void* d_ws, size_t ws_size,
                              hipStream_t stream) {
    const float* tF   = (const float*)d_in[0];
    const float* sF   = (const float*)d_in[1];
    const float* tw1  = (const float*)d_in[2];
    const float* tb1  = (const float*)d_in[3];
    const float* tw2  = (const float*)d_in[4];
    const float* tb2  = (const float*)d_in[5];
    const float* tg   = (const float*)d_in[6];
    const float* tbe  = (const float*)d_in[7];
    const float* sw1  = (const float*)d_in[8];
    const float* sb1  = (const float*)d_in[9];
    const float* sw2  = (const float*)d_in[10];
    const float* sb2  = (const float*)d_in[11];
    const float* sg   = (const float*)d_in[12];
    const float* sbe  = (const float*)d_in[13];
    const int*   tcrd = (const int*)d_in[14];
    const int*   scrd = (const int*)d_in[15];
    float* out = (float*)d_out;

    char* ws = (char*)d_ws;
    size_t off = 0;
    auto alloc = [&](size_t bytes) -> char* {
        char* p = ws + off;
        off = (off + bytes + 255) & ~(size_t)255;
        return p;
    };
    const size_t NF = (size_t)NTEACH * FD;
    bf16_t* tF_bf    = (bf16_t*)alloc(NF * 2);
    bf16_t* sF_bf    = (bf16_t*)alloc(NF * 2);
    bf16_t* tw1t_bf  = (bf16_t*)alloc(FD * FD * 2);   // transposed (N-major) weights
    bf16_t* tw2t_bf  = (bf16_t*)alloc(FD * FD * 2);
    bf16_t* sw1t_bf  = (bf16_t*)alloc(FD * FD * 2);
    bf16_t* sw2t_bf  = (bf16_t*)alloc(FD * FD * 2);
    bf16_t* h1t_bf   = (bf16_t*)alloc(NF * 2);
    bf16_t* h1s_bf   = (bf16_t*)alloc(NF * 2);
    float*  h2t      = (float*)alloc(NF * 4);
    float*  h2s      = (float*)alloc(NF * 4);
    float*  mu_t     = (float*)alloc(FD * 4);
    float*  var_t    = (float*)alloc(FD * 4);
    float*  mu_s     = (float*)alloc(FD * 4);
    float*  var_s    = (float*)alloc(FD * 4);
    float*  tproj    = (float*)alloc(NF * 4);
    float*  sproj    = (float*)alloc(NF * 4);
    bf16_t* tproj_bf = (bf16_t*)alloc(NF * 2);
    bf16_t* sproj_bf = (bf16_t*)alloc(NF * 2);
    int*    tkeys    = (int*)alloc(NTEACH * 4);
    float*  pos_raw  = (float*)alloc(NSTUD * 4);
    float*  mflag    = (float*)alloc(NSTUD * 4);
    float*  accum    = (float*)alloc(2 * 4);

    k_init_accum<<<1, 32, 0, stream>>>(accum);

    int nelem = (int)NF;
    k_cvt_bf16<<<(nelem + 255) / 256, 256, 0, stream>>>(tF, tF_bf, nelem);
    k_cvt_bf16<<<(nelem + 255) / 256, 256, 0, stream>>>(sF, sF_bf, nelem);
    k_cvt_bf16_t<<<(FD * FD + 255) / 256, 256, 0, stream>>>(tw1, tw1t_bf, FD, FD);
    k_cvt_bf16_t<<<(FD * FD + 255) / 256, 256, 0, stream>>>(tw2, tw2t_bf, FD, FD);
    k_cvt_bf16_t<<<(FD * FD + 255) / 256, 256, 0, stream>>>(sw1, sw1t_bf, FD, FD);
    k_cvt_bf16_t<<<(FD * FD + 255) / 256, 256, 0, stream>>>(sw2, sw2t_bf, FD, FD);

    // MLP layer 1 (ReLU, bf16 out) and layer 2 (f32 out)
    k_gemm_mlp<true,  false><<<NTEACH / 16, 256, 0, stream>>>(tF_bf, tw1t_bf, tb1, nullptr, h1t_bf);
    k_gemm_mlp<true,  false><<<NSTUD  / 16, 256, 0, stream>>>(sF_bf, sw1t_bf, sb1, nullptr, h1s_bf);
    k_gemm_mlp<false, true ><<<NTEACH / 16, 256, 0, stream>>>(h1t_bf, tw2t_bf, tb2, h2t, nullptr);
    k_gemm_mlp<false, true ><<<NSTUD  / 16, 256, 0, stream>>>(h1s_bf, sw2t_bf, sb2, h2s, nullptr);

    k_bn_stats<<<FD, 256, 0, stream>>>(h2t, mu_t, var_t, NTEACH);
    k_bn_stats<<<FD, 256, 0, stream>>>(h2s, mu_s, var_s, NSTUD);
    k_bn_norm<<<NTEACH, 128, 0, stream>>>(h2t, mu_t, var_t, tg, tbe, tproj, tproj_bf);
    k_bn_norm<<<NSTUD, 128, 0, stream>>>(h2s, mu_s, var_s, sg, sbe, sproj, sproj_bf);

    k_keys<<<(NTEACH + 255) / 256, 256, 0, stream>>>(tcrd, tkeys, NTEACH);
    k_match<<<(NSTUD + 255) / 256, 256, 0, stream>>>(scrd, tkeys, sproj, tproj,
                                                     pos_raw, mflag, accum);

    k_topk_lse<<<NSTUD / 64, 256, 0, stream>>>(sproj_bf, tproj_bf, pos_raw, mflag, accum);

    k_finalize<<<1, 32, 0, stream>>>(accum, out);
}
